// AutoRegressionModel_58119497450397
// MI455X (gfx1250) — compile-verified, matrix operand order checked by reading, not run on hardware
//
#include <hip/hip_runtime.h>

typedef __attribute__((ext_vector_type(16))) _Float16 v16h;
typedef __attribute__((ext_vector_type(8)))  _Float16 h8;
typedef __attribute__((ext_vector_type(8)))  float    v8f;

#define LSTM_B 256
#define LSTM_T 512
#define LSTM_D 768
#define LSTM_H 128
#define G4H    512
#define K1_STRIDE 40   // padded LDS row stride (halves): 80B rows -> conflict-free frag reads
#define NKT (LSTM_D / 32)

__device__ __forceinline__ v16h mk_frag(h8 lo, h8 hi) {
  v16h v;
#pragma unroll
  for (int i = 0; i < 8; ++i) { v[i] = lo[i]; v[i + 8] = hi[i]; }
  return v;
}

__device__ __forceinline__ float sigm(float x) { return 1.0f / (1.0f + __expf(-x)); }
__device__ __forceinline__ float tanh_fast(float x) { return 2.0f * sigm(2.0f * x) - 1.0f; }

// ---------------------------------------------------------------------------
// Phase 1: xg[t][b][4H] = x[b][t][:] @ W_ih^T + (b_ih + b_hh)
// Workgroup = 128M x 128N, 8 waves (4 in M x 2 in N), wave = 32M x 64N.
// Double-buffered LDS (one barrier per K-step) + register prefetch of the
// next K-tile so HBM latency hides behind barrier + WMMAs.
// ---------------------------------------------------------------------------
__global__ __launch_bounds__(256)
void xg_gemm_kernel(const float* __restrict__ x, const float* __restrict__ W_ih,
                    const float* __restrict__ b_ih, const float* __restrict__ b_hh,
                    float* __restrict__ xg) {
  __shared__ __align__(16) _Float16 Abuf[2][128 * K1_STRIDE];
  __shared__ __align__(16) _Float16 Bbuf[2][128 * K1_STRIDE];

  const int tid   = threadIdx.x;
  const int lane  = tid & 31;
  const int wave  = tid >> 5;
  const int r16   = lane & 15;
  const int halfl = lane >> 4;

  const int m0 = blockIdx.x * 128;
  const int n0 = blockIdx.y * 128;
  const int wm = wave >> 1;   // 0..3 -> 32-row strip
  const int wn = wave & 1;    // 0..1 -> 64-col strip

  v8f acc[2][4];
#pragma unroll
  for (int i = 0; i < 2; ++i)
#pragma unroll
    for (int j = 0; j < 4; ++j)
#pragma unroll
      for (int r = 0; r < 8; ++r) acc[i][j][r] = 0.0f;

  const int srow  = tid >> 1;   // staging: 2 threads per row, 16 floats each
  const int shalf = tid & 1;
  const float* aptr = x    + (size_t)(m0 + srow) * LSTM_D + shalf * 16;
  const float* bptr = W_ih + (size_t)(n0 + srow) * LSTM_D + shalf * 16;

  float4 av[4], bv[4];
#pragma unroll
  for (int q = 0; q < 4; ++q) {
    av[q] = ((const float4*)aptr)[q];
    bv[q] = ((const float4*)bptr)[q];
  }

  for (int kk = 0; kk < NKT; ++kk) {
    const int p = kk & 1;
    {
      _Float16* adst = &Abuf[p][srow * K1_STRIDE + shalf * 16];
      _Float16* bdst = &Bbuf[p][srow * K1_STRIDE + shalf * 16];
#pragma unroll
      for (int q = 0; q < 4; ++q) {
        adst[q * 4 + 0] = (_Float16)av[q].x; adst[q * 4 + 1] = (_Float16)av[q].y;
        adst[q * 4 + 2] = (_Float16)av[q].z; adst[q * 4 + 3] = (_Float16)av[q].w;
        bdst[q * 4 + 0] = (_Float16)bv[q].x; bdst[q * 4 + 1] = (_Float16)bv[q].y;
        bdst[q * 4 + 2] = (_Float16)bv[q].z; bdst[q * 4 + 3] = (_Float16)bv[q].w;
      }
    }
    if (kk + 1 < NKT) {  // prefetch next K-tile; consumed next iteration
      const float* an = aptr + (kk + 1) * 32;
      const float* bn = bptr + (kk + 1) * 32;
#pragma unroll
      for (int q = 0; q < 4; ++q) {
        av[q] = ((const float4*)an)[q];
        bv[q] = ((const float4*)bn)[q];
      }
    }
    __syncthreads();   // single barrier per step (double-buffered LDS)

    v16h afr[2], bfr[4];
#pragma unroll
    for (int mt = 0; mt < 2; ++mt) {
      const _Float16* pp = &Abuf[p][(wm * 32 + mt * 16 + r16) * K1_STRIDE + halfl * 8];
      afr[mt] = mk_frag(*(const h8*)pp, *(const h8*)(pp + 16));
    }
#pragma unroll
    for (int nt = 0; nt < 4; ++nt) {
      const _Float16* pp = &Bbuf[p][(wn * 64 + nt * 16 + r16) * K1_STRIDE + halfl * 8];
      bfr[nt] = mk_frag(*(const h8*)pp, *(const h8*)(pp + 16));
    }
#pragma unroll
    for (int mt = 0; mt < 2; ++mt)
#pragma unroll
      for (int nt = 0; nt < 4; ++nt)
        acc[mt][nt] = __builtin_amdgcn_wmma_f32_16x16x32_f16(
            false, afr[mt], false, bfr[nt], (short)0, acc[mt][nt], false, false);
  }

  // epilogue: + (b_ih + b_hh), scatter GEMM row m = b*T + t -> xg[t][b][col]
#pragma unroll
  for (int mt = 0; mt < 2; ++mt)
#pragma unroll
    for (int nt = 0; nt < 4; ++nt) {
      const int col  = n0 + wn * 64 + nt * 16 + r16;
      const float bias = b_ih[col] + b_hh[col];
#pragma unroll
      for (int r = 0; r < 8; ++r) {
        const int m  = m0 + wm * 32 + mt * 16 + halfl * 8 + r;
        const int bi = m >> 9;        // batch
        const int ti = m & 511;       // timestep
        xg[((size_t)ti * LSTM_B + bi) * G4H + col] = acc[mt][nt][r] + bias;
      }
    }
}

// ---------------------------------------------------------------------------
// Phase 2: sequential LSTM (512 steps) + MLP head.
// 16 workgroups x 16-batch-row tiles (batch rows independent -> no grid sync).
// Wave w owns hidden units [16w,16w+16), all four gates -> i/f/g/o element-
// aligned in accumulators; c lives in accumulator registers; W_hh^T fragments
// live in registers for all 512 steps. Double-buffered h in LDS (one barrier
// per step) and software-pipelined xg prefetch (t+1 loads fly during WMMAs).
// ---------------------------------------------------------------------------
__global__ __launch_bounds__(256)
void lstm_head_kernel(const float* __restrict__ xg, const float* __restrict__ W_hh,
                      const float* __restrict__ w1, const float* __restrict__ b1,
                      const float* __restrict__ w2, const float* __restrict__ b2,
                      const float* __restrict__ w3, const float* __restrict__ b3,
                      float* __restrict__ out) {
  __shared__ __align__(16) _Float16 hbuf[2][16 * LSTM_H];
  __shared__ float hf[16 * LSTM_H];
  __shared__ float z1s[16 * 64];
  __shared__ float z2s[16 * 32];

  const int tid   = threadIdx.x;
  const int lane  = tid & 31;
  const int wave  = tid >> 5;   // 0..7
  const int r16   = lane & 15;
  const int halfl = lane >> 4;
  const int b0    = blockIdx.x * 16;
  const int u0    = wave * 16;  // hidden-unit slice

  // h0 = 0 (buffer 0 only; buffer 1 is written at t=0 before being read)
#pragma unroll
  for (int j = 0; j < 8; ++j) hbuf[0][tid * 8 + j] = (_Float16)0.0f;

  // Preload B = W_hh^T fragments (constant over time): B(k',n') = W_hh[G*128+u0+n'][k']
  v16h bfr[4][4];
#pragma unroll
  for (int G = 0; G < 4; ++G)
#pragma unroll
    for (int k = 0; k < 4; ++k) {
      const float* wp = W_hh + (size_t)(G * LSTM_H + u0 + r16) * LSTM_H + k * 32 + halfl * 8;
      v16h v;
#pragma unroll
      for (int j = 0; j < 8; ++j) { v[j] = (_Float16)wp[j]; v[j + 8] = (_Float16)wp[16 + j]; }
      bfr[G][k] = v;
    }

  auto load_xg = [&](int t, v8f pf[4]) {
    const float* xgp = xg + ((size_t)t * LSTM_B + b0) * G4H;
#pragma unroll
    for (int G = 0; G < 4; ++G) {
      const int col = G * LSTM_H + u0 + r16;
#pragma unroll
      for (int r = 0; r < 8; ++r)
        pf[G][r] = xgp[(size_t)(halfl * 8 + r) * G4H + col];
    }
  };

  v8f cacc;
  float hval[8];
#pragma unroll
  for (int r = 0; r < 8; ++r) { cacc[r] = 0.0f; hval[r] = 0.0f; }

  v8f pref[4];
  load_xg(0, pref);

  for (int t = 0; t < LSTM_T; ++t) {
    __syncthreads();   // step t-1's h stores (or init) visible; prior reads done
    const int p = t & 1;

    v8f acc[4];
#pragma unroll
    for (int G = 0; G < 4; ++G) acc[G] = pref[G];   // C init = xg[t] (+bias), prefetched
    if (t + 1 < LSTM_T) load_xg(t + 1, pref);       // in flight during WMMAs

#pragma unroll
    for (int k = 0; k < 4; ++k) {
      const _Float16* pp = &hbuf[p][r16 * LSTM_H + k * 32 + halfl * 8];
      const v16h a = mk_frag(*(const h8*)pp, *(const h8*)(pp + 16));
#pragma unroll
      for (int G = 0; G < 4; ++G)
        acc[G] = __builtin_amdgcn_wmma_f32_16x16x32_f16(
            false, a, false, bfr[G][k], (short)0, acc[G], false, false);
    }

#pragma unroll
    for (int r = 0; r < 8; ++r) {
      const float iv = sigm(acc[0][r]);
      const float fv = sigm(acc[1][r]);
      const float gv = tanh_fast(acc[2][r]);
      const float ov = sigm(acc[3][r]);
      const float cv = fv * cacc[r] + iv * gv;
      cacc[r] = cv;
      hval[r] = ov * tanh_fast(cv);
    }

    // write h(t+1 input) into the other buffer; next top barrier publishes it
#pragma unroll
    for (int r = 0; r < 8; ++r)
      hbuf[p ^ 1][(halfl * 8 + r) * LSTM_H + u0 + r16] = (_Float16)hval[r];
  }

  // publish final h (fp32) for the head
#pragma unroll
  for (int r = 0; r < 8; ++r)
    hf[(halfl * 8 + r) * LSTM_H + u0 + r16] = hval[r];
  __syncthreads();

  // z1 = relu(h @ w1^T + b1): 16 rows x 64, 4 outputs/thread
  {
    const int row = tid >> 4, j4 = tid & 15;
#pragma unroll
    for (int jj = 0; jj < 4; ++jj) {
      const int j = j4 * 4 + jj;
      float s = b1[j];
      for (int d = 0; d < LSTM_H; ++d) s += hf[row * LSTM_H + d] * w1[j * LSTM_H + d];
      z1s[row * 64 + j] = fmaxf(s, 0.0f);
    }
  }
  __syncthreads();
  // z2 = relu(z1 @ w2^T + b2): 16 x 32, 2 outputs/thread
  {
    const int row = tid >> 4, j2 = tid & 15;
#pragma unroll
    for (int jj = 0; jj < 2; ++jj) {
      const int j = j2 * 2 + jj;
      float s = b2[j];
      for (int d = 0; d < 64; ++d) s += z1s[row * 64 + d] * w2[j * 64 + d];
      z2s[row * 32 + j] = fmaxf(s, 0.0f);
    }
  }
  __syncthreads();
  // out = sigmoid(z2 @ w3^T + b3): one thread per batch row
  if (tid < 16) {
    float s = b3[0];
    for (int d = 0; d < 32; ++d) s += z2s[tid * 32 + d] * w3[d];
    out[b0 + tid] = sigm(s);
  }
}

extern "C" void kernel_launch(void* const* d_in, const int* in_sizes, int n_in,
                              void* d_out, int out_size, void* d_ws, size_t ws_size,
                              hipStream_t stream) {
  const float* x    = (const float*)d_in[0];
  const float* W_ih = (const float*)d_in[1];
  const float* W_hh = (const float*)d_in[2];
  const float* b_ih = (const float*)d_in[3];
  const float* b_hh = (const float*)d_in[4];
  const float* w1   = (const float*)d_in[5];
  const float* b1   = (const float*)d_in[6];
  const float* w2   = (const float*)d_in[7];
  const float* b2   = (const float*)d_in[8];
  const float* w3   = (const float*)d_in[9];
  const float* b3   = (const float*)d_in[10];

  float* xg = (float*)d_ws;   // 512*256*512 fp32 = 256 MB workspace: xg[t][b][4H]

  dim3 g1((LSTM_B * LSTM_T) / 128, G4H / 128);     // 1024 x 4 workgroups
  xg_gemm_kernel<<<g1, 256, 0, stream>>>(x, W_ih, b_ih, b_hh, xg);
  lstm_head_kernel<<<LSTM_B / 16, 256, 0, stream>>>(xg, W_hh, w1, b1, w2, b2, w3, b3,
                                                    (float*)d_out);
  (void)in_sizes; (void)n_in; (void)out_size; (void)ws_size;
}
